// FullSymmetryBrokenAttention_38534446580116
// MI455X (gfx1250) — matrix operand order repaired and checked by reading, not compiled, not run on hardware
//
#include <hip/hip_runtime.h>
#include <hip/hip_bf16.h>

typedef __attribute__((ext_vector_type(16))) _Float16 v16h;
typedef __attribute__((ext_vector_type(8)))  _Float16 v8h;
typedef __attribute__((ext_vector_type(2)))  _Float16 h2_t;
typedef __attribute__((ext_vector_type(8)))  float    v8f;
typedef __attribute__((ext_vector_type(2)))  float    f2_t;

#define NH   16
#define TSEQ 2048
#define CDIM 1024
#define DH   64
#define BATCH 4

static __device__ __forceinline__ v8f zero_v8f() {
  v8f z = {0.f,0.f,0.f,0.f,0.f,0.f,0.f,0.f};
  return z;
}

static __device__ __forceinline__ v8f wmma_f16(v16h a, v16h b, v8f c) {
  // D = A(16x32 f16) * B(32x16 f16) + C(16x16 f32)
  return __builtin_amdgcn_wmma_f32_16x16x32_f16(false, a, false, b, (short)0, c, false, false);
}

// Base K of pair p (p=0..7) inside a 16x32 f16 A-fragment, for half-wave hi.
// Contiguous order: half-wave hi holds K {0..15}+hi*16, matching the B fragments.
static __device__ __forceinline__ int a_kpair(int hi, int p) {
  return hi * 16 + (p << 1);
}

union AFragU { v16h v; _Float16 h[16]; h2_t p[8]; };
union BFragU { v16h v; _Float16 h[16]; v8h  q[2]; };

// ---------------------------------------------------------------------------
// Shared 128x128 WMMA GEMM core. 256 threads = 8 waves in 4(M) x 2(N) grid;
// each wave computes a 32x64 tile = 2x4 wmma accumulators. LDS tiles are kept
// in fragment-swizzled order: [block 0..7][lane 0..31][16 halves], so each
// fragment load is one aligned 32-byte LDS read.
// ---------------------------------------------------------------------------
template<bool A_IS_F16>
static __device__ __forceinline__ void gemm_128x128(
    const void* __restrict__ Aptr, const float* __restrict__ Bptr,
    int lda, int ldb, int Ktot, int m0, int n0,
    _Float16* lA, _Float16* lB, v8f acc[2][4])
{
  const int t    = threadIdx.x;
  const int lane = t & 31;
  const int blk  = t >> 5;          // 0..7: stage-block / wave id
  const int hi   = lane >> 4;
  const int wm   = blk & 3;
  const int wn   = blk >> 2;
  const int mrow = (blk << 4) + (lane & 15);   // staged A row 0..127
  const int ncol = (blk << 4) + (lane & 15);   // staged B col 0..127

  #pragma unroll
  for (int i = 0; i < 2; ++i)
    #pragma unroll
    for (int j = 0; j < 4; ++j) acc[i][j] = zero_v8f();

  for (int k0 = 0; k0 < Ktot; k0 += 32) {
    // ---- stage A tile (128 x 32) into per-lane fragment order (f32->f16) ----
    {
      _Float16* dst = &lA[(blk * 32 + lane) * 16];
      if (A_IS_F16) {
        const _Float16* arow = (const _Float16*)Aptr + (size_t)(m0 + mrow) * lda + k0;
        if (k0 + 32 < Ktot) __builtin_prefetch(arow + 32, 0, 0);
        #pragma unroll
        for (int p = 0; p < 8; ++p) {
          const int kk = a_kpair(hi, p);
          *(h2_t*)&dst[2 * p] = *(const h2_t*)&arow[kk];
        }
      } else {
        const float* arow = (const float*)Aptr + (size_t)(m0 + mrow) * lda + k0;
        if (k0 + 32 < Ktot) __builtin_prefetch(arow + 32, 0, 0);
        #pragma unroll
        for (int p = 0; p < 8; ++p) {
          const int kk = a_kpair(hi, p);
          f2_t xv = *(const f2_t*)&arow[kk];
          h2_t hv; hv[0] = (_Float16)xv[0]; hv[1] = (_Float16)xv[1];
          *(h2_t*)&dst[2 * p] = hv;
        }
      }
    }
    // ---- stage B tile (32 x 128): lane = column, 16 consecutive K per half --
    {
      const float* bsrc = Bptr + (size_t)(k0 + hi * 16) * ldb + (n0 + ncol);
      BFragU tmp;
      #pragma unroll
      for (int e = 0; e < 16; ++e) tmp.h[e] = (_Float16)bsrc[(size_t)e * ldb];
      _Float16* dst = &lB[(blk * 32 + lane) * 16];
      *(v8h*)&dst[0] = tmp.q[0];
      *(v8h*)&dst[8] = tmp.q[1];
    }
    __syncthreads();

    AFragU af[2]; BFragU bf[4];
    #pragma unroll
    for (int i = 0; i < 2; ++i)
      af[i].v = *(const v16h*)&lA[(((wm << 1) + i) * 32 + lane) * 16];
    #pragma unroll
    for (int j = 0; j < 4; ++j)
      bf[j].v = *(const v16h*)&lB[(((wn << 2) + j) * 32 + lane) * 16];
    #pragma unroll
    for (int i = 0; i < 2; ++i)
      #pragma unroll
      for (int j = 0; j < 4; ++j)
        acc[i][j] = wmma_f16(af[i].v, bf[j].v, acc[i][j]);
    __syncthreads();
  }
}

// ---------------------------------------------------------------------------
// Kernel 1: qkv = x @ W_attn + b_attn, fused per-head biases, f16 outputs.
// Q,K in [b*nh][t][d]; V TRANSPOSED to [b*nh][d][t] for contiguous B-frags.
// ---------------------------------------------------------------------------
__global__ __launch_bounds__(256)
void qkv_gemm_kernel(const float* __restrict__ x, const float* __restrict__ W,
                     const float* __restrict__ b_attn,
                     const float* __restrict__ bQ, const float* __restrict__ bK,
                     const float* __restrict__ bV,
                     _Float16* __restrict__ qg, _Float16* __restrict__ kg,
                     _Float16* __restrict__ vg)
{
  __shared__ __align__(32) _Float16 lA[8 * 32 * 16];
  __shared__ __align__(32) _Float16 lB[8 * 32 * 16];
  const int m0 = blockIdx.y * 128;
  const int n0 = blockIdx.x * 128;
  v8f acc[2][4];
  gemm_128x128<false>(x, W, CDIM, 3 * CDIM, CDIM, m0, n0, lA, lB, acc);

  const int lane = threadIdx.x & 31;
  const int blk  = threadIdx.x >> 5;
  const int hi   = lane >> 4;
  const int ln15 = lane & 15;
  const int wm   = blk & 3, wn = blk >> 2;

  #pragma unroll
  for (int i = 0; i < 2; ++i)
    #pragma unroll
    for (int j = 0; j < 4; ++j)
      #pragma unroll
      for (int ii = 0; ii < 8; ++ii) {
        const int m = m0 + ((wm << 1) + i) * 16 + ii + (hi << 3);
        const int n = n0 + ((wn << 2) + j) * 16 + ln15;
        float val = acc[i][j][ii] + b_attn[n];
        const int sec = n >> 10;        // 0=q 1=k 2=v
        const int c   = n & 1023;
        const int h   = c >> 6;
        const int dc  = c & 63;
        const int b   = m >> 11;        // / TSEQ
        const int tt  = m & 2047;
        const size_t bh = (size_t)(b * NH + h);
        if (sec == 0) {
          qg[(bh * TSEQ + tt) * DH + dc] = (_Float16)(val + bQ[h * DH + dc]);
        } else if (sec == 1) {
          kg[(bh * TSEQ + tt) * DH + dc] = (_Float16)(val + bK[h * DH + dc]);
        } else {
          vg[(bh * DH + dc) * TSEQ + tt] = (_Float16)(val + bV[h * DH + dc]);
        }
      }
}

// ---------------------------------------------------------------------------
// Kernel 2: causal flash attention. One 16-query tile per wave, 32-key blocks.
// ---------------------------------------------------------------------------
__global__ __launch_bounds__(128)
void attn_kernel(const _Float16* __restrict__ qg, const _Float16* __restrict__ kg,
                 const _Float16* __restrict__ vg, _Float16* __restrict__ yg)
{
  __shared__ __align__(32) _Float16 pbuf[4][16 * 32];   // per-wave P scratch
  const int lane = threadIdx.x & 31;
  const int wid  = threadIdx.x >> 5;
  const int hi   = lane >> 4;
  const int ln15 = lane & 15;
  const int tile = blockIdx.x * 4 + wid;   // 0..8191
  const int qt   = tile & (TSEQ / 16 - 1); // 128 q-tiles per (b,h)
  const int bh   = tile >> 7;
  const int q0   = qt << 4;

  // Q fragments (two 16x32 d-chunks), prescaled by 1/sqrt(64)
  AFragU aq[2];
  {
    const _Float16* qrow = qg + ((size_t)bh * TSEQ + q0 + ln15) * DH;
    #pragma unroll
    for (int cch = 0; cch < 2; ++cch)
      #pragma unroll
      for (int p = 0; p < 8; ++p) {
        const int kk = cch * 32 + a_kpair(hi, p);
        h2_t hv = *(const h2_t*)&qrow[kk];
        hv[0] = hv[0] * (_Float16)0.125f;
        hv[1] = hv[1] * (_Float16)0.125f;
        aq[cch].p[p] = hv;
      }
  }

  v8f o[4];
  #pragma unroll
  for (int j = 0; j < 4; ++j) o[j] = zero_v8f();
  float mrow[8], lrow[8];
  #pragma unroll
  for (int ii = 0; ii < 8; ++ii) { mrow[ii] = -3.0e38f; lrow[ii] = 0.f; }

  const int nkb = (q0 >> 5) + 1;       // key blocks of 32 covering [0, q0+16)
  _Float16* pb = &pbuf[wid][0];

  for (int kb = 0; kb < nkb; ++kb) {
    const int key0 = kb << 5;
    // ---- scores: S(16x32) via 2 key-tiles x 2 d-chunks = 4 WMMAs ----
    v8f s[2];
    #pragma unroll
    for (int kt = 0; kt < 2; ++kt) {
      const _Float16* krow =
          kg + ((size_t)bh * TSEQ + key0 + kt * 16 + ln15) * DH + hi * 16;
      v16h b0 = *(const v16h*)&krow[0];    // d = 0..31 half for this lane
      v16h b1 = *(const v16h*)&krow[32];   // d = 32..63
      v8f cc = zero_v8f();
      cc    = wmma_f16(aq[0].v, b0, cc);
      s[kt] = wmma_f16(aq[1].v, b1, cc);
    }
    // ---- causal mask on the diagonal block ----
    if (kb == nkb - 1) {
      #pragma unroll
      for (int kt = 0; kt < 2; ++kt)
        #pragma unroll
        for (int ii = 0; ii < 8; ++ii) {
          const int qm = q0 + ii + (hi << 3);
          const int kn = key0 + kt * 16 + ln15;
          if (kn > qm) s[kt][ii] = -3.0e38f;
        }
    }
    // ---- online softmax (row lives in one 16-lane half; C-layout) ----
    #pragma unroll
    for (int ii = 0; ii < 8; ++ii) {
      float mx = fmaxf(s[0][ii], s[1][ii]);
      #pragma unroll
      for (int off = 8; off >= 1; off >>= 1) mx = fmaxf(mx, __shfl_xor(mx, off, 16));
      const float mnew  = fmaxf(mrow[ii], mx);
      const float alpha = __expf(mrow[ii] - mnew);
      const float p0 = __expf(s[0][ii] - mnew);
      const float p1 = __expf(s[1][ii] - mnew);
      float rs = p0 + p1;
      #pragma unroll
      for (int off = 8; off >= 1; off >>= 1) rs += __shfl_xor(rs, off, 16);
      lrow[ii] = lrow[ii] * alpha + rs;
      mrow[ii] = mnew;
      s[0][ii] = p0; s[1][ii] = p1;
      #pragma unroll
      for (int j = 0; j < 4; ++j) o[j][ii] *= alpha;
    }
    // ---- re-layout P (C-layout -> A-fragment) through per-wave LDS ----
    #pragma unroll
    for (int kt = 0; kt < 2; ++kt)
      #pragma unroll
      for (int ii = 0; ii < 8; ++ii)
        pb[(ii + (hi << 3)) * 32 + kt * 16 + ln15] = (_Float16)s[kt][ii];
    asm volatile("s_wait_dscnt 0" ::: "memory");
    AFragU ap;
    #pragma unroll
    for (int p = 0; p < 8; ++p) {
      const int kk = a_kpair(hi, p);
      ap.p[p] = *(const h2_t*)&pb[ln15 * 32 + kk];
    }
    asm volatile("" ::: "memory");   // keep loads before next-iter overwrite
    // ---- O += P @ V  (V transposed: contiguous 16 keys per lane) ----
    #pragma unroll
    for (int j = 0; j < 4; ++j) {
      v16h bv = *(const v16h*)
          &vg[((size_t)bh * DH + j * 16 + ln15) * TSEQ + key0 + hi * 16];
      o[j] = wmma_f16(ap.v, bv, o[j]);
    }
  }

  // ---- normalize and write y back in [b*T][C] f16 for the proj GEMM ----
  const int b = bh >> 4, h = bh & 15;
  #pragma unroll
  for (int ii = 0; ii < 8; ++ii) {
    const float inv = 1.0f / lrow[ii];
    const size_t row = (size_t)(b * TSEQ + q0 + ii + (hi << 3));
    #pragma unroll
    for (int j = 0; j < 4; ++j)
      yg[row * CDIM + h * DH + j * 16 + ln15] = (_Float16)(o[j][ii] * inv);
  }
}

// ---------------------------------------------------------------------------
// Kernel 3: out = y @ W_proj + b_proj  (f32 output)
// ---------------------------------------------------------------------------
__global__ __launch_bounds__(256)
void proj_gemm_kernel(const _Float16* __restrict__ y, const float* __restrict__ W,
                      const float* __restrict__ b_proj, float* __restrict__ out)
{
  __shared__ __align__(32) _Float16 lA[8 * 32 * 16];
  __shared__ __align__(32) _Float16 lB[8 * 32 * 16];
  const int m0 = blockIdx.y * 128;
  const int n0 = blockIdx.x * 128;
  v8f acc[2][4];
  gemm_128x128<true>(y, W, CDIM, CDIM, CDIM, m0, n0, lA, lB, acc);

  const int lane = threadIdx.x & 31;
  const int blk  = threadIdx.x >> 5;
  const int hi   = lane >> 4;
  const int ln15 = lane & 15;
  const int wm   = blk & 3, wn = blk >> 2;
  #pragma unroll
  for (int i = 0; i < 2; ++i)
    #pragma unroll
    for (int j = 0; j < 4; ++j)
      #pragma unroll
      for (int ii = 0; ii < 8; ++ii) {
        const int m = m0 + ((wm << 1) + i) * 16 + ii + (hi << 3);
        const int n = n0 + ((wn << 2) + j) * 16 + ln15;
        out[(size_t)m * CDIM + n] = acc[i][j][ii] + b_proj[n];
      }
}

// ---------------------------------------------------------------------------
extern "C" void kernel_launch(void* const* d_in, const int* in_sizes, int n_in,
                              void* d_out, int out_size, void* d_ws, size_t ws_size,
                              hipStream_t stream) {
  const float* x      = (const float*)d_in[0];
  const float* W_attn = (const float*)d_in[1];
  const float* b_attn = (const float*)d_in[2];
  const float* W_proj = (const float*)d_in[3];
  const float* b_proj = (const float*)d_in[4];
  const float* bQ     = (const float*)d_in[5];
  const float* bK     = (const float*)d_in[6];
  const float* bV     = (const float*)d_in[7];
  float* out = (float*)d_out;

  const size_t elems = (size_t)BATCH * NH * TSEQ * DH;   // 8,388,608 halves
  _Float16* qg = (_Float16*)d_ws;
  _Float16* kg = qg + elems;
  _Float16* vg = kg + elems;
  _Float16* yg = vg + elems;   // [B*T][C] f16, same element count

  dim3 blk256(256);
  dim3 g1(3 * CDIM / 128, (BATCH * TSEQ) / 128);   // 24 x 64
  qkv_gemm_kernel<<<g1, blk256, 0, stream>>>(x, W_attn, b_attn, bQ, bK, bV,
                                             qg, kg, vg);

  dim3 g2((BATCH * NH * (TSEQ / 16)) / 4);          // 2048 blocks x 4 waves
  attn_kernel<<<g2, dim3(128), 0, stream>>>(qg, kg, vg, yg);

  dim3 g3(CDIM / 128, (BATCH * TSEQ) / 128);        // 8 x 64
  proj_gemm_kernel<<<g3, blk256, 0, stream>>>(yg, W_proj, b_proj, out);
}
